// SparseAutoencoder_64647847740069
// MI455X (gfx1250) — matrix operand order repaired and checked by reading, not compiled, not run on hardware
//
#include <hip/hip_runtime.h>
#include <hip/hip_bf16.h>
#include <stdint.h>

// ---------------------------------------------------------------------------
// TopK Sparse Autoencoder forward for MI455X (gfx1250, wave32, WMMA + async LDS)
//   outputs (concat): recon [8192,1024] | sparse_acts [8192,16384] | pre_acts [8192,16384]
// ---------------------------------------------------------------------------

#define DM    1024      // d_model
#define DH    16384     // d_hidden
#define NTOK  8192      // n_tokens
#define KSEL  32        // top-k

typedef __bf16 bf16;
typedef __attribute__((ext_vector_type(16))) __bf16 v16bf;
typedef __attribute__((ext_vector_type(8)))  float  v8f;
typedef __attribute__((ext_vector_type(4)))  float  v4f;

// ---------------------------------------------------------------------------
// Kernel 0: fold b_pre into the activation bias:
//   bias2[h] = b_act[h] - dot(b_pre, Wenc[h,:])      (exact f32)
// so the encode GEMM can consume raw x tiles via async global->LDS staging.
// ---------------------------------------------------------------------------
__global__ __launch_bounds__(256)
void fold_bias(const float* __restrict__ Wenc, const float* __restrict__ b_pre,
               const float* __restrict__ b_act, float* __restrict__ bias2)
{
    const int h = blockIdx.x;
    __shared__ float red[256];
    const float* w = Wenc + (size_t)h * DM;
    float s = 0.0f;
    for (int d = threadIdx.x; d < DM; d += 256) s += b_pre[d] * w[d];
    red[threadIdx.x] = s;
    __syncthreads();
    for (int off = 128; off > 0; off >>= 1) {
        if (threadIdx.x < off) red[threadIdx.x] += red[threadIdx.x + off];
        __syncthreads();
    }
    if (threadIdx.x == 0) bias2[h] = b_act[h] - red[0];
}

// ---------------------------------------------------------------------------
// Kernel 1: encode GEMM  pre = x @ Wenc^T + bias2   (split-bf16 WMMA, f32 acc)
//           also zero-fills sparse_acts.
// Block tile 128(M) x 128(H), BK=32. 256 threads = 8 waves, each wave 32x64.
// Tiles staged as raw f32 via GLOBAL_LOAD_ASYNC_TO_LDS_B128 (ASYNCcnt);
// bf16 hi/lo split happens at fragment build (co-executes with WMMA).
// ---------------------------------------------------------------------------
#define BM 128
#define BN 128
#define BK 32
#define LDF 36   // f32 LDS row stride: 144B rows (16B aligned), conflict-free

__global__ __launch_bounds__(256)
void encode_gemm(const float* __restrict__ x, const float* __restrict__ Wenc,
                 const float* __restrict__ bias2,
                 float* __restrict__ pre_out, float* __restrict__ sparse_out)
{
    __shared__ alignas(16) float sA[BM][LDF];
    __shared__ alignas(16) float sB[BN][LDF];

    const int tid  = threadIdx.x;
    const int wave = tid >> 5;
    const int lane = tid & 31;
    const int m0   = blockIdx.y * BM;
    const int h0   = blockIdx.x * BN;
    const int wm   = (wave >> 1) * 32;   // 4 waves along M
    const int wn   = (wave & 1) * 64;    // 2 waves along N

    v8f acc[2][4] = {};

    const int lrow = tid >> 3;           // 0..31
    const int lcol = (tid & 7) * 4;      // 0..28 step 4

    for (int kc = 0; kc < DM; kc += BK) {
        // ---- async-stage x and Wenc f32 tiles straight into LDS ----
        #pragma unroll
        for (int rr = 0; rr < 4; ++rr) {
            const int r = lrow + rr * 32;
            const uint64_t ga = (uint64_t)(uintptr_t)&x[(size_t)(m0 + r) * DM + kc + lcol];
            const uint32_t la = (uint32_t)(uintptr_t)&sA[r][lcol];
            asm volatile("global_load_async_to_lds_b128 %0, %1, off"
                         :: "v"(la), "v"(ga) : "memory");
            const uint64_t gb = (uint64_t)(uintptr_t)&Wenc[(size_t)(h0 + r) * DM + kc + lcol];
            const uint32_t lb = (uint32_t)(uintptr_t)&sB[r][lcol];
            asm volatile("global_load_async_to_lds_b128 %0, %1, off"
                         :: "v"(lb), "v"(gb) : "memory");
        }
        asm volatile("s_wait_asynccnt 0x0" ::: "memory");
        __syncthreads();

        // ---- fragment build (ISA 7.12.2 16-bit A 16x32 / B 32x16, wave32) ----
        const int fr  = lane & 15;
        const int aks = (lane < 16) ? 0 : 8;   // A half-lane K base (v0-3), +16 for v4-7
        const int bks = (lane < 16) ? 0 : 8;   // B half-lane K base (v0-3), +16 for v4-7

        v16bf Ah[2], Al[2];
        #pragma unroll
        for (int i = 0; i < 2; ++i) {
            const float* pa = &sA[wm + i * 16 + fr][0];
            float f[16];
            *(v4f*)&f[0]  = *(const v4f*)(pa + aks);
            *(v4f*)&f[4]  = *(const v4f*)(pa + aks + 4);
            *(v4f*)&f[8]  = *(const v4f*)(pa + aks + 16);
            *(v4f*)&f[12] = *(const v4f*)(pa + aks + 20);
            #pragma unroll
            for (int e = 0; e < 16; ++e) {
                bf16 hv = (bf16)f[e];
                Ah[i][e] = hv;
                Al[i][e] = (bf16)(f[e] - (float)hv);
            }
        }

        #pragma unroll
        for (int j = 0; j < 4; ++j) {
            const float* pb = &sB[wn + j * 16 + fr][0];
            float g[16];
            *(v4f*)&g[0]  = *(const v4f*)(pb + bks);
            *(v4f*)&g[4]  = *(const v4f*)(pb + bks + 4);
            *(v4f*)&g[8]  = *(const v4f*)(pb + bks + 16);
            *(v4f*)&g[12] = *(const v4f*)(pb + bks + 20);
            v16bf Bh, Bl;
            #pragma unroll
            for (int e = 0; e < 16; ++e) {
                bf16 hv = (bf16)g[e];
                Bh[e] = hv;
                Bl[e] = (bf16)(g[e] - (float)hv);
            }
            // D = Ah*Bh + Ah*Bl + Al*Bh  (split-bf16 ~ f32 accuracy)
            #pragma unroll
            for (int i = 0; i < 2; ++i) {
                acc[i][j] = __builtin_amdgcn_wmma_f32_16x16x32_bf16(
                    false, Ah[i], false, Bh, (short)0, acc[i][j], false, false);
                acc[i][j] = __builtin_amdgcn_wmma_f32_16x16x32_bf16(
                    false, Ah[i], false, Bl, (short)0, acc[i][j], false, false);
                acc[i][j] = __builtin_amdgcn_wmma_f32_16x16x32_bf16(
                    false, Al[i], false, Bh, (short)0, acc[i][j], false, false);
            }
        }
        __syncthreads();
    }

    // Epilogue: C/D layout — VGPR v: lanes0-15 row M=v, lanes16-31 row M=v+8
    const int col  = lane & 15;
    const int rofs = (lane < 16) ? 0 : 8;
    #pragma unroll
    for (int i = 0; i < 2; ++i)
        #pragma unroll
        for (int j = 0; j < 4; ++j) {
            const int   hcol = h0 + wn + j * 16 + col;
            const float ba   = bias2[hcol];
            #pragma unroll
            for (int v = 0; v < 8; ++v) {
                const int    row = m0 + wm + i * 16 + rofs + v;
                const size_t off = (size_t)row * DH + hcol;
                pre_out[off]    = acc[i][j][v] + ba;
                sparse_out[off] = 0.0f;
            }
        }
}

// ---------------------------------------------------------------------------
// Kernel 2: per-row top-k via 4-pass MSB radix select; scatter relu vals.
// ---------------------------------------------------------------------------
__device__ __forceinline__ unsigned tokey(float v) {
    unsigned u = __float_as_uint(v);
    return (u & 0x80000000u) ? ~u : (u | 0x80000000u);   // monotone total order
}

__global__ __launch_bounds__(256)
void topk_select(const float* __restrict__ pre, float* __restrict__ sparse_out,
                 int* __restrict__ idx_out, float* __restrict__ val_out)
{
    const int row = blockIdx.x;
    const float* p = pre + (size_t)row * DH;
    __shared__ unsigned hist[256];
    __shared__ unsigned s_prefix;
    __shared__ int s_remaining, s_cntG, s_cntT;

    const int tid = threadIdx.x;
    if (tid == 0) { s_prefix = 0; s_remaining = KSEL; }
    __syncthreads();

    for (int pass = 0; pass < 4; ++pass) {
        const int shift = 24 - 8 * pass;
        hist[tid] = 0;
        __syncthreads();
        const unsigned pref  = s_prefix;
        const unsigned hmask = (pass == 0) ? 0u : (0xFFFFFFFFu << (shift + 8));
        for (int i = tid; i < DH; i += 256) {
            unsigned key = tokey(p[i]);
            if ((key & hmask) == (pref & hmask))
                atomicAdd(&hist[(key >> shift) & 255u], 1u);
        }
        __syncthreads();
        if (tid == 0) {
            int rem = s_remaining;
            unsigned b = 255;
            for (;; --b) {
                int c = (int)hist[b];
                if (rem <= c || b == 0) break;
                rem -= c;
            }
            s_prefix    = pref | (b << shift);
            s_remaining = rem;                 // ties of this key to take
        }
        __syncthreads();
    }

    const unsigned T = s_prefix;
    const int      r = s_remaining;
    if (tid == 0) { s_cntG = 0; s_cntT = 0; }
    __syncthreads();

    for (int i = tid; i < DH; i += 256) {
        const float    v   = p[i];
        const unsigned key = tokey(v);
        int slot = -1;
        if (key > T) {
            slot = atomicAdd(&s_cntG, 1);
        } else if (key == T) {
            int t = atomicAdd(&s_cntT, 1);
            if (t < r) slot = (KSEL - r) + t;
        }
        if (slot >= 0) {
            const float vr = v > 0.0f ? v : 0.0f;          // hard = relu(topk)
            sparse_out[(size_t)row * DH + i] = vr;
            idx_out[row * KSEL + slot] = i;
            val_out[row * KSEL + slot] = vr;
        }
    }
}

// ---------------------------------------------------------------------------
// Kernel 3: transpose W_dec [D,H] -> W_decT [H,D]  (decode gathers become rows)
// ---------------------------------------------------------------------------
__global__ __launch_bounds__(256)
void transpose_wdec(const float* __restrict__ Wdec, float* __restrict__ WdecT)
{
    __shared__ float tile[32][33];
    const int hb = blockIdx.x * 32;
    const int db = blockIdx.y * 32;
    const int tx = threadIdx.x & 31;
    const int ty = threadIdx.x >> 5;
    #pragma unroll
    for (int i = 0; i < 32; i += 8)
        tile[ty + i][tx] = Wdec[(size_t)(db + ty + i) * DH + hb + tx];
    __syncthreads();
    #pragma unroll
    for (int i = 0; i < 32; i += 8)
        WdecT[(size_t)(hb + ty + i) * DM + db + tx] = tile[tx][ty + i];
}

// ---------------------------------------------------------------------------
// Kernel 4: sparse decode  recon[n,:] = b_pre + sum_j val_j * W_decT[idx_j,:]
// W_decT (64MB) is L2-resident (192MB L2) -> gathers mostly hit L2.
// ---------------------------------------------------------------------------
__global__ __launch_bounds__(256)
void decode_sparse(const float* __restrict__ WdecT, const float* __restrict__ b_pre,
                   const int* __restrict__ idx_in, const float* __restrict__ val_in,
                   float* __restrict__ recon)
{
    const int row = blockIdx.x;
    __shared__ int   sidx[KSEL];
    __shared__ float sval[KSEL];
    const int tid = threadIdx.x;
    if (tid < KSEL) {
        sidx[tid] = idx_in[row * KSEL + tid];
        sval[tid] = val_in[row * KSEL + tid];
    }
    __syncthreads();

    const int d = tid * 4;                 // 256 threads * float4 = 1024
    float4 acc = *(const float4*)&b_pre[d];
    #pragma unroll 8
    for (int j = 0; j < KSEL; ++j) {
        const float4 w = *(const float4*)&WdecT[(size_t)sidx[j] * DM + d];
        const float  s = sval[j];
        acc.x += s * w.x; acc.y += s * w.y; acc.z += s * w.z; acc.w += s * w.w;
    }
    *(float4*)&recon[(size_t)row * DM + d] = acc;
}

// ---------------------------------------------------------------------------
extern "C" void kernel_launch(void* const* d_in, const int* in_sizes, int n_in,
                              void* d_out, int out_size, void* d_ws, size_t ws_size,
                              hipStream_t stream)
{
    const float* x     = (const float*)d_in[0];
    const float* Wenc  = (const float*)d_in[1];
    const float* Wdec  = (const float*)d_in[2];
    const float* b_pre = (const float*)d_in[3];
    const float* b_act = (const float*)d_in[4];
    // d_in[5] is k (==32), compiled in as KSEL.

    float* out    = (float*)d_out;
    float* recon  = out;                                   // [NTOK, DM]
    float* sparse = out + (size_t)NTOK * DM;               // [NTOK, DH]
    float* pre    = sparse + (size_t)NTOK * DH;            // [NTOK, DH]

    // workspace: W_decT (64MB) | idx list (1MB) | val list (1MB) | bias2 (64KB)
    float* WdecT = (float*)d_ws;
    int*   idxL  = (int*)  ((char*)d_ws + (size_t)DH * DM * sizeof(float));
    float* valL  = (float*)((char*)idxL + (size_t)NTOK * KSEL * sizeof(int));
    float* bias2 = (float*)((char*)valL + (size_t)NTOK * KSEL * sizeof(float));

    fold_bias    <<<DH, 256, 0, stream>>>(Wenc, b_pre, b_act, bias2);
    transpose_wdec<<<dim3(DH / 32, DM / 32), 256, 0, stream>>>(Wdec, WdecT);
    encode_gemm  <<<dim3(DH / BN, NTOK / BM), 256, 0, stream>>>(x, Wenc, bias2, pre, sparse);
    topk_select  <<<NTOK, 256, 0, stream>>>(pre, sparse, idxL, valL);
    decode_sparse<<<NTOK, 256, 0, stream>>>(WdecT, b_pre, idxL, valL, recon);
}
